// GroupedQueryAttention_74655121539486
// MI455X (gfx1250) — matrix operand order repaired and checked by reading, not compiled, not run on hardware
//
#include <hip/hip_runtime.h>

// ---------------- CDNA5 WMMA types / fragment helpers ----------------
typedef __attribute__((ext_vector_type(16))) _Float16 v16h;
typedef __attribute__((ext_vector_type(8)))  float    v8f;

union FragH {            // 16x32 f16 A-operand or 32x16 f16 B-operand
    v16h      v;
    _Float16  h[16];
    unsigned  u[8];
    uint4     q[2];
};
union FragF {            // 16x16 f32 C/D
    v8f   v;
    float f[8];
};

static __device__ __forceinline__ v8f wmma_f16(const v16h& a, const v16h& b, const v8f& c) {
    return __builtin_amdgcn_wmma_f32_16x16x32_f16(false, a, false, b, (short)0, c, false, false);
}

// Async global->LDS copy of 64 bytes (4x B128), CDNA5 ASYNCcnt path.
// INST_OFFSET is added to BOTH the LDS and global addresses (ISA 08 §4.4),
// so one (lds,vaddr) pair covers the whole 64B run via offset immediates.
static __device__ __forceinline__ void async_copy_64B(unsigned lds_addr,
                                                      const _Float16* gaddr) {
    asm volatile("global_load_async_to_lds_b128 %0, %1, off\n\t"
                 "global_load_async_to_lds_b128 %0, %1, off offset:16\n\t"
                 "global_load_async_to_lds_b128 %0, %1, off offset:32\n\t"
                 "global_load_async_to_lds_b128 %0, %1, off offset:48"
                 :: "v"(lds_addr), "v"(gaddr) : "memory");
}
static __device__ __forceinline__ void wait_async0() {
    asm volatile("s_wait_asynccnt 0x0" ::: "memory");
}

// ---------------- fp32 -> fp16 conversion (row-major copy) ----------------
__global__ void cvt_f32_f16(const float* __restrict__ in, _Float16* __restrict__ out, int n) {
    int i = blockIdx.x * blockDim.x + threadIdx.x;
    if (i < n) out[i] = (_Float16)in[i];
}

// ---------------- fp32 (KxN) -> fp16 transposed (NxK), tiled ----------------
__global__ __launch_bounds__(256) void transpose_cvt(const float* __restrict__ in,
                                                     _Float16* __restrict__ out,
                                                     int K, int N) {
    __shared__ _Float16 tile[64][65];
    const int k0 = blockIdx.y * 64;
    const int n0 = blockIdx.x * 64;
#pragma unroll
    for (int r = 0; r < 16; ++r) {
        int idx = r * 256 + threadIdx.x;    // 0..4095
        int kk = idx >> 6, nn = idx & 63;
        tile[nn][kk] = (_Float16)in[(size_t)(k0 + kk) * N + n0 + nn];
    }
    __syncthreads();
#pragma unroll
    for (int r = 0; r < 16; ++r) {
        int idx = r * 256 + threadIdx.x;
        int nn = idx >> 6, kk = idx & 63;
        out[(size_t)(n0 + nn) * K + k0 + kk] = tile[nn][kk];
    }
}

// ---------------- Tiled WMMA GEMM:  C(MxN) = A(MxK) @ Bt(NxK)^T + bias ----------
// A f16 row-major; Bt f16 TRANSPOSED (N x K) so a B-operand fragment is a
// contiguous 32-byte run per lane, streamed from global/L2.
// 256 threads = 8 waves. Block tile: 128(M) x 128(N), K-step 64.
// A tiles are DOUBLE-BUFFERED in LDS via async global->LDS copies (ASYNCcnt),
// overlapping HBM traffic with the 16 WMMAs/K-step each wave issues.
template <typename OutT>
__global__ __launch_bounds__(256) void gemm_wmma(const _Float16* __restrict__ A,
                                                 const _Float16* __restrict__ Bt,
                                                 const float* __restrict__ bias,
                                                 OutT* __restrict__ C,
                                                 int M, int N, int K) {
    // [buf][m][k]; row stride 144B keeps every ds b128 access 16B-aligned
    __shared__ _Float16 As[2][128][72];

    const int t    = threadIdx.x;
    const int lane = t & 31;
    const int wave = t >> 5;               // 0..7
    const int lrow = lane & 15;
    const int hi   = (lane >= 16);
    const int m0   = blockIdx.y * 128;
    const int n0   = blockIdx.x * 128;
    const int n    = n0 + wave * 16 + lrow;

    FragF acc[8];
#pragma unroll
    for (int mc = 0; mc < 8; ++mc)
#pragma unroll
        for (int j = 0; j < 8; ++j) acc[mc].f[j] = 0.0f;

    const _Float16* brow = &Bt[(size_t)n * K];

    // per-thread slice of the A tile: 64B at (row = t/2, c0 = (t&1)*32)
    const int arow = t >> 1;
    const int ac0  = (t & 1) * 32;
    const _Float16* ag = &A[(size_t)(m0 + arow) * K + ac0];

    // prologue: stage tile 0 into buffer 0
    async_copy_64B((unsigned)(size_t)&As[0][arow][ac0], ag);
    wait_async0();
    __syncthreads();

    int buf = 0;
    for (int k0 = 0; k0 < K; k0 += 64) {
        // kick off async staging of the next tile into the other buffer
        if (k0 + 64 < K)
            async_copy_64B((unsigned)(size_t)&As[buf ^ 1][arow][ac0], ag + k0 + 64);

        // prefetch next B run while the matrix pipes chew on this tile
        __builtin_prefetch(brow + k0 + 64, 0, 1);

#pragma unroll
        for (int kc = 0; kc < 2; ++kc) {
            // B fragment straight from global: contiguous 32B per lane
            FragH fb;
            const _Float16* bp = brow + k0 + kc * 32 + (hi ? 16 : 0);
            fb.q[0] = *reinterpret_cast<const uint4*>(bp);
            fb.q[1] = *reinterpret_cast<const uint4*>(bp + 8);
#pragma unroll
            for (int mc = 0; mc < 8; ++mc) {
                // A fragment from LDS: two 16B runs (K {0-15,16-31} per lane half)
                FragH fa;
                const _Float16* ap = &As[buf][mc * 16 + lrow][kc * 32 + (hi ? 16 : 0)];
                fa.q[0] = *reinterpret_cast<const uint4*>(ap);
                fa.q[1] = *reinterpret_cast<const uint4*>(ap + 8);
                acc[mc].v = wmma_f16(fa.v, fb.v, acc[mc].v);
            }
        }
        wait_async0();     // my async writes into buf^1 are done
        __syncthreads();   // everyone finished reading buf + staging buf^1
        buf ^= 1;
    }

    const float bv = bias[n];
#pragma unroll
    for (int mc = 0; mc < 8; ++mc)
#pragma unroll
        for (int j = 0; j < 8; ++j) {
            int m = m0 + mc * 16 + j + (hi ? 8 : 0);
            C[(size_t)m * N + n] = (OutT)(acc[mc].f[j] + bv);
        }
}

// ---------------- Flash-attention (GQA) with WMMA ----------------
// Q: (B*S, 2048) f16 row-major (head h at cols h*64..).  K,V: (B*S, 512) f16.
// Grid: x = B*H (b=idx/32, h=idx%32), y = S/64.  Block: 128 thr = 4 waves,
// each wave owns 16 query rows.  Key tiles of 32.
__global__ __launch_bounds__(128) void attn_wmma(const _Float16* __restrict__ Qb,
                                                 const _Float16* __restrict__ Kb,
                                                 const _Float16* __restrict__ Vb,
                                                 _Float16* __restrict__ Ob) {
    constexpr int S = 2048, D = 2048, KVD = 512, HD = 64;
    constexpr float scale = 0.125f;  // 1/sqrt(64)

    __shared__ _Float16 Vs[64][34];      // transposed V tile: [hd][key]
    __shared__ _Float16 Ps[4][16][36];   // per-wave P tile (16x32)

    const int bh   = blockIdx.x;
    const int b    = bh >> 5;
    const int h    = bh & 31;
    const int kvh  = h >> 2;             // GQA: 4 query heads per kv head
    const int q0   = blockIdx.y * 64;
    const int t    = threadIdx.x;
    const int lane = t & 31;
    const int wave = t >> 5;             // 0..3
    const int lrow = lane & 15;
    const int hi   = (lane >= 16);

    // ---- Q fragments (A layout), head_dim 64 -> two 16x32 fragments ----
    FragH fq[2];
    {
        const int qrow = q0 + wave * 16 + lrow;
        const size_t qoff = (size_t)(b * S + qrow) * D + h * HD;
#pragma unroll
        for (int f = 0; f < 2; ++f)
#pragma unroll
            for (int p = 0; p < 8; ++p) {
                int kd = f * 32 + 2 * p + (hi ? 16 : 0);
                fq[f].u[p] = *reinterpret_cast<const unsigned*>(&Qb[qoff + kd]);
            }
    }

    FragF acc[4];
#pragma unroll
    for (int n = 0; n < 4; ++n)
#pragma unroll
        for (int j = 0; j < 8; ++j) acc[n].f[j] = 0.0f;
    float mprev[8], lsum[8];
#pragma unroll
    for (int j = 0; j < 8; ++j) { mprev[j] = -1e30f; lsum[j] = 0.0f; }

    for (int jt = 0; jt < S / 32; ++jt) {
        const int jj = jt * 32;

        // ---- stage V tile (32 keys x 64 hd) transposed into LDS ----
        {
            int keyl = t >> 2;            // 0..31
            int hd0  = (t & 3) * 16;      // 0..48
            const _Float16* vp =
                &Vb[(size_t)(b * S + jj + keyl) * KVD + kvh * HD + hd0];
            uint4 a0 = *reinterpret_cast<const uint4*>(vp);
            uint4 a1 = *reinterpret_cast<const uint4*>(vp + 8);
            const _Float16* h0 = reinterpret_cast<const _Float16*>(&a0);
            const _Float16* h1 = reinterpret_cast<const _Float16*>(&a1);
#pragma unroll
            for (int j = 0; j < 8; ++j) Vs[hd0 + j][keyl] = h0[j];
#pragma unroll
            for (int j = 0; j < 8; ++j) Vs[hd0 + 8 + j][keyl] = h1[j];
        }

        // ---- scores: S(16x32) = Q(16x64) . K^T(64x32), two 16x16 C tiles ----
        FragF s0, s1;
#pragma unroll
        for (int j = 0; j < 8; ++j) { s0.f[j] = 0.0f; s1.f[j] = 0.0f; }
#pragma unroll
        for (int st = 0; st < 2; ++st) {
            const int key = jj + st * 16 + lrow;   // B-operand column = key
            FragF* sc = st ? &s1 : &s0;
#pragma unroll
            for (int kc = 0; kc < 2; ++kc) {
                // K^T column == contiguous K row: 32B straight from global
                const _Float16* kp = &Kb[(size_t)(b * S + key) * KVD +
                                         kvh * HD + kc * 32 + (hi ? 16 : 0)];
                FragH fk;
                fk.q[0] = *reinterpret_cast<const uint4*>(kp);
                fk.q[1] = *reinterpret_cast<const uint4*>(kp + 8);
                sc->v = wmma_f16(fq[kc].v, fk.v, sc->v);
            }
        }

        // ---- online softmax (rows live across 16-lane half-groups) ----
        float p0[8], p1[8], alpha[8];
#pragma unroll
        for (int j = 0; j < 8; ++j) {
            s0.f[j] *= scale; s1.f[j] *= scale;
            float mrow = fmaxf(s0.f[j], s1.f[j]);
#pragma unroll
            for (int msk = 1; msk < 16; msk <<= 1)
                mrow = fmaxf(mrow, __shfl_xor(mrow, msk, 32));
            float mn = fmaxf(mprev[j], mrow);
            alpha[j] = __expf(mprev[j] - mn);
            p0[j] = __expf(s0.f[j] - mn);
            p1[j] = __expf(s1.f[j] - mn);
            float rs = p0[j] + p1[j];
#pragma unroll
            for (int msk = 1; msk < 16; msk <<= 1)
                rs += __shfl_xor(rs, msk, 32);
            lsum[j] = lsum[j] * alpha[j] + rs;
            mprev[j] = mn;
        }
#pragma unroll
        for (int n = 0; n < 4; ++n)
#pragma unroll
            for (int j = 0; j < 8; ++j) acc[n].f[j] *= alpha[j];

        // ---- stage P (C layout -> LDS) then reload as A fragment ----
#pragma unroll
        for (int j = 0; j < 8; ++j) {
            int row = j + (hi ? 8 : 0);
            Ps[wave][row][lrow]      = (_Float16)p0[j];
            Ps[wave][row][16 + lrow] = (_Float16)p1[j];
        }
        __syncthreads();

        FragH fp;
#pragma unroll
        for (int p = 0; p < 8; ++p) {
            int kk = 2 * p + (hi ? 16 : 0);
            fp.u[p] = *reinterpret_cast<const unsigned*>(&Ps[wave][lrow][kk]);
        }

        // ---- O(16x64) += P(16x32) . V(32x64): 4 WMMAs over hd chunks ----
#pragma unroll
        for (int n = 0; n < 4; ++n) {
            const _Float16* vrow = &Vs[n * 16 + lrow][hi ? 16 : 0];
            FragH fv;
#pragma unroll
            for (int p = 0; p < 8; ++p)
                fv.u[p] = *reinterpret_cast<const unsigned*>(vrow + 2 * p);
            acc[n].v = wmma_f16(fp.v, fv.v, acc[n].v);
        }
        __syncthreads();
    }

    // ---- normalize and write attention output (f16, row-major MxD) ----
#pragma unroll
    for (int n = 0; n < 4; ++n) {
#pragma unroll
        for (int j = 0; j < 8; ++j) {
            int q  = q0 + wave * 16 + j + (hi ? 8 : 0);
            int hd = n * 16 + lrow;
            Ob[(size_t)(b * S + q) * D + h * HD + hd] =
                (_Float16)(acc[n].f[j] / lsum[j]);
        }
    }
}

// ---------------- host-side launch ----------------
extern "C" void kernel_launch(void* const* d_in, const int* in_sizes, int n_in,
                              void* d_out, int out_size, void* d_ws, size_t ws_size,
                              hipStream_t stream) {
    (void)in_sizes; (void)n_in; (void)out_size; (void)ws_size;

    constexpr int B = 2, S = 2048, D = 2048, H = 32, KVD = 512;
    constexpr int M = B * S;                       // 4096 rows

    const float* x  = (const float*)d_in[0];
    const float* Wq = (const float*)d_in[1];
    const float* bq = (const float*)d_in[2];
    const float* Wk = (const float*)d_in[3];
    const float* bk = (const float*)d_in[4];
    const float* Wv = (const float*)d_in[5];
    const float* bv = (const float*)d_in[6];
    const float* Wo = (const float*)d_in[7];
    const float* bo = (const float*)d_in[8];
    float* out = (float*)d_out;

    // workspace carve-up (element counts, f16)
    _Float16* w = (_Float16*)d_ws;
    const size_t nx  = (size_t)M * D;     // 8.39M
    const size_t nwq = (size_t)D * D;     // 4.19M
    const size_t nwk = (size_t)D * KVD;   // 1.05M
    _Float16* xh  = w;            w += nx;
    _Float16* Wqt = w;            w += nwq;   // transposed (N x K)
    _Float16* Wkt = w;            w += nwk;
    _Float16* Wvt = w;            w += nwk;
    _Float16* Wot = w;            w += nwq;
    _Float16* Qh  = w;            w += nx;
    _Float16* Kh  = w;            w += (size_t)M * KVD;
    _Float16* Vh  = w;            w += (size_t)M * KVD;
    _Float16* Ah  = w;            w += nx;

    // activations: plain convert; weights: transpose-convert (K x N -> N x K)
    cvt_f32_f16<<<(unsigned)((nx + 255) / 256), 256, 0, stream>>>(x, xh, (int)nx);
    transpose_cvt<<<dim3(D / 64,   D / 64), 256, 0, stream>>>(Wq, Wqt, D, D);
    transpose_cvt<<<dim3(KVD / 64, D / 64), 256, 0, stream>>>(Wk, Wkt, D, KVD);
    transpose_cvt<<<dim3(KVD / 64, D / 64), 256, 0, stream>>>(Wv, Wvt, D, KVD);
    transpose_cvt<<<dim3(D / 64,   D / 64), 256, 0, stream>>>(Wo, Wot, D, D);

    // projections (block tile 128x128, async double-buffered A tiles)
    gemm_wmma<_Float16><<<dim3(D / 128,   M / 128), 256, 0, stream>>>(xh, Wqt, bq, Qh, M, D,   D);
    gemm_wmma<_Float16><<<dim3(KVD / 128, M / 128), 256, 0, stream>>>(xh, Wkt, bk, Kh, M, KVD, D);
    gemm_wmma<_Float16><<<dim3(KVD / 128, M / 128), 256, 0, stream>>>(xh, Wvt, bv, Vh, M, KVD, D);

    // attention: (B*H) x (S/64) blocks, 128 threads
    attn_wmma<<<dim3(B * H, S / 64), 128, 0, stream>>>(Qh, Kh, Vh, Ah);

    // output projection -> fp32
    gemm_wmma<float><<<dim3(D / 128, M / 128), 256, 0, stream>>>(Ah, Wot, bo, out, M, D, D);
}
